// EdgeMLP_16080357556865
// MI455X (gfx1250) — matrix operand order repaired and checked by reading, not compiled, not run on hardware
//
#include <hip/hip_runtime.h>
#include <math.h>

typedef __attribute__((ext_vector_type(16))) __bf16 v16bf;
typedef __attribute__((ext_vector_type(8)))  __bf16 v8bf;
typedef __attribute__((ext_vector_type(4)))  __bf16 v4bf;
typedef __attribute__((ext_vector_type(8)))  float  v8f;

#define HDIM 128
#define DIN 384
#define TM 64            // edges per block
#define SA_STRIDE 392    // 384 + 8 pad (bf16 elems) -> 784B row, 196 DW % 64 = 4 (conflict-free b128)
#define SH_STRIDE 136    // 128 + 8 pad -> 272B row, 68 DW % 64 = 4
#define SMEM_BYTES (TM*SA_STRIDE*2 + 1024)
#define H2_OFF (TM*SH_STRIDE*2)        // 17408
#define RED_OFF (TM*SA_STRIDE*2)       // 50176

// d_ws byte layout: [0,512) colsum, [512,1024) colsumsq, [1024,1536) scale,
// [1536,2048) shift, [2048, ...) packed bf16 weights (W11 49152 | W12 16384 | W13 16384 elems)
#define W12_OFF 49152
#define W13_OFF 65536
#define WTOTAL  81920

__device__ __forceinline__ float gelu_exact(float x) {
    return 0.5f * x * (1.0f + erff(x * 0.70710678118654752f));
}

// Pack f32 weights into bf16 B-operand fragment layout:
// fragment (kbIdx,nbIdx) = 512 elems; elem d: lane = (d>>4)&31, j = d&15
// h = lane>>4: K = kb + (j<8 ? 8h + j : 16 + 8h + j-8), N = nb + (lane&15)
__global__ __launch_bounds__(256) void pack_weights(const float* __restrict__ W11,
                                                    const float* __restrict__ W12,
                                                    const float* __restrict__ W13,
                                                    __bf16* __restrict__ wp) {
    int i = blockIdx.x * blockDim.x + threadIdx.x;
    if (i >= WTOTAL) return;
    const float* W; int d;
    if (i < W12_OFF)      { W = W11; d = i; }
    else if (i < W13_OFF) { W = W12; d = i - W12_OFF; }
    else                  { W = W13; d = i - W13_OFF; }
    int frag = d >> 9;
    int r = d & 511;
    int lane = r >> 4;
    int j = r & 15;
    int kbIdx = frag >> 3;
    int nbIdx = frag & 7;
    int K = kbIdx * 32 + ((j >> 3) << 4) + ((lane >> 4) << 3) + (j & 7);
    int N = nbIdx * 16 + (lane & 15);
    wp[i] = (__bf16)W[K * HDIM + N];
}

__global__ __launch_bounds__(256) void edge_mlp(
    const float* __restrict__ src_na, const float* __restrict__ dst_na,
    const float* __restrict__ ea, const int* __restrict__ edge_idx,
    const float* __restrict__ b11, const float* __restrict__ b12,
    const float* __restrict__ b13, const __bf16* __restrict__ wp,
    float* __restrict__ out, float* __restrict__ gsum, float* __restrict__ gsq, int E)
{
    __shared__ __align__(16) unsigned char smem[SMEM_BYTES];
    __bf16* sA  = (__bf16*)smem;
    __bf16* sH1 = (__bf16*)smem;              // overlay (barrier-separated)
    __bf16* sH2 = (__bf16*)(smem + H2_OFF);   // overlay
    float*  redS = (float*)(smem + RED_OFF);
    float*  redQ = (float*)(smem + RED_OFF + 512);

    const int tid  = threadIdx.x;
    const int lane = tid & 31;
    const int wave = tid >> 5;
    const int e0   = blockIdx.x * TM;

    if (tid < 128) { redS[tid] = 0.f; redQ[tid] = 0.f; }

    // ---- gather + f32->bf16 convert: rows of [src_na | ea | dst_na] ----
    for (int i = tid; i < TM * 96; i += 256) {
        int row = i / 96;
        int c4  = i - row * 96;          // col = c4*4
        int e = e0 + row; if (e >= E) e = E - 1;
        int seg = c4 >> 5;               // 0: src, 1: ea, 2: dst
        int inner = c4 & 31;
        const float4* src;
        if (seg == 0)      { int s = edge_idx[E + e]; src = (const float4*)(src_na + (size_t)s * HDIM) + inner; }
        else if (seg == 1) {               src = (const float4*)(ea     + (size_t)e * HDIM) + inner; }
        else               { int dn = edge_idx[e];  src = (const float4*)(dst_na + (size_t)dn * HDIM) + inner; }
        float4 f = *src;
        v4bf hv; hv[0] = (__bf16)f.x; hv[1] = (__bf16)f.y; hv[2] = (__bf16)f.z; hv[3] = (__bf16)f.w;
        *(v4bf*)(sA + row * SA_STRIDE + c4 * 4) = hv;
    }
    __syncthreads();

    const int mtile  = wave >> 1;        // 0..3, 16 rows each
    const int nhalf  = wave & 1;         // n-tiles nhalf*4 .. +3
    const int lhalf8 = (lane >> 4) * 8;  // A-layout K sub-block
    const int nlane  = lane & 15;

    v8f acc[4];

    // ================= GEMM1: [64x384] x W11 =================
#pragma unroll
    for (int t = 0; t < 4; ++t) {
        float bv = b11[(nhalf * 4 + t) * 16 + nlane];
#pragma unroll
        for (int i = 0; i < 8; ++i) acc[t][i] = bv;
    }
    {
        const __bf16* arow = sA + (mtile * 16 + nlane) * SA_STRIDE;
        for (int kb = 0; kb < DIN; kb += 32) {
            v8bf alo = *(const v8bf*)(arow + kb + lhalf8);
            v8bf ahi = *(const v8bf*)(arow + kb + 16 + lhalf8);
            v16bf a = __builtin_shufflevector(alo, ahi, 0,1,2,3,4,5,6,7,8,9,10,11,12,13,14,15);
            const __bf16* wb = wp + ((kb >> 5) * 8) * 512 + lane * 16;
#pragma unroll
            for (int t = 0; t < 4; ++t) {
                v16bf b = *(const v16bf*)(wb + (nhalf * 4 + t) * 512);
                acc[t] = __builtin_amdgcn_wmma_f32_16x16x32_bf16(
                    false, a, false, b, (short)0, acc[t], false, false);
            }
        }
    }
    __syncthreads();   // all GEMM1 reads of sA done before h1 overlay writes
#pragma unroll
    for (int t = 0; t < 4; ++t) {
        int col = (nhalf * 4 + t) * 16 + nlane;
#pragma unroll
        for (int r = 0; r < 8; ++r) {
            int row = mtile * 16 + r + lhalf8;
            sH1[row * SH_STRIDE + col] = (__bf16)gelu_exact(acc[t][r]);
        }
    }
    __syncthreads();

    // ================= GEMM2: h1 x W12 =================
#pragma unroll
    for (int t = 0; t < 4; ++t) {
        float bv = b12[(nhalf * 4 + t) * 16 + nlane];
#pragma unroll
        for (int i = 0; i < 8; ++i) acc[t][i] = bv;
    }
    {
        const __bf16* arow = sH1 + (mtile * 16 + nlane) * SH_STRIDE;
        const __bf16* wbase = wp + W12_OFF;
        for (int kb = 0; kb < HDIM; kb += 32) {
            v8bf alo = *(const v8bf*)(arow + kb + lhalf8);
            v8bf ahi = *(const v8bf*)(arow + kb + 16 + lhalf8);
            v16bf a = __builtin_shufflevector(alo, ahi, 0,1,2,3,4,5,6,7,8,9,10,11,12,13,14,15);
            const __bf16* wb = wbase + ((kb >> 5) * 8) * 512 + lane * 16;
#pragma unroll
            for (int t = 0; t < 4; ++t) {
                v16bf b = *(const v16bf*)(wb + (nhalf * 4 + t) * 512);
                acc[t] = __builtin_amdgcn_wmma_f32_16x16x32_bf16(
                    false, a, false, b, (short)0, acc[t], false, false);
            }
        }
    }
    // write h2 to disjoint overlay region; barrier before GEMM3 reads
#pragma unroll
    for (int t = 0; t < 4; ++t) {
        int col = (nhalf * 4 + t) * 16 + nlane;
#pragma unroll
        for (int r = 0; r < 8; ++r) {
            int row = mtile * 16 + r + lhalf8;
            sH2[row * SH_STRIDE + col] = (__bf16)gelu_exact(acc[t][r]);
        }
    }
    __syncthreads();

    // ================= GEMM3: h2 x W13 =================
#pragma unroll
    for (int t = 0; t < 4; ++t) {
        float bv = b13[(nhalf * 4 + t) * 16 + nlane];
#pragma unroll
        for (int i = 0; i < 8; ++i) acc[t][i] = bv;
    }
    {
        const __bf16* arow = sH2 + (mtile * 16 + nlane) * SH_STRIDE;
        const __bf16* wbase = wp + W13_OFF;
        for (int kb = 0; kb < HDIM; kb += 32) {
            v8bf alo = *(const v8bf*)(arow + kb + lhalf8);
            v8bf ahi = *(const v8bf*)(arow + kb + 16 + lhalf8);
            v16bf a = __builtin_shufflevector(alo, ahi, 0,1,2,3,4,5,6,7,8,9,10,11,12,13,14,15);
            const __bf16* wb = wbase + ((kb >> 5) * 8) * 512 + lane * 16;
#pragma unroll
            for (int t = 0; t < 4; ++t) {
                v16bf b = *(const v16bf*)(wb + (nhalf * 4 + t) * 512);
                acc[t] = __builtin_amdgcn_wmma_f32_16x16x32_bf16(
                    false, a, false, b, (short)0, acc[t], false, false);
            }
        }
    }

    // ---- x = ea + msg, write out, per-column partial sums ----
#pragma unroll
    for (int t = 0; t < 4; ++t) {
        int col = (nhalf * 4 + t) * 16 + nlane;
        float s = 0.f, q = 0.f;
#pragma unroll
        for (int r = 0; r < 8; ++r) {
            int el = mtile * 16 + r + lhalf8;
            int e  = e0 + el;
            if (e < E) {
                float x = acc[t][r] + ea[(size_t)e * HDIM + col];
                out[(size_t)e * HDIM + col] = x;
                s += x; q += x * x;
            }
        }
        s += __shfl_xor(s, 16, 32);
        q += __shfl_xor(q, 16, 32);
        if (lane < 16) {
            __hip_atomic_fetch_add(&redS[col], s, __ATOMIC_RELAXED, __HIP_MEMORY_SCOPE_WORKGROUP);
            __hip_atomic_fetch_add(&redQ[col], q, __ATOMIC_RELAXED, __HIP_MEMORY_SCOPE_WORKGROUP);
        }
    }
    __syncthreads();
    if (tid < 128) {
        __hip_atomic_fetch_add(&gsum[tid], redS[tid], __ATOMIC_RELAXED, __HIP_MEMORY_SCOPE_AGENT);
        __hip_atomic_fetch_add(&gsq[tid],  redQ[tid], __ATOMIC_RELAXED, __HIP_MEMORY_SCOPE_AGENT);
    }
}

__global__ void bn_finalize(const float* __restrict__ gsum, const float* __restrict__ gsq,
                            const float* __restrict__ gamma, const float* __restrict__ beta,
                            float* __restrict__ scale, float* __restrict__ shift, int E) {
    int c = threadIdx.x;
    if (c < HDIM) {
        float invE = 1.0f / (float)E;
        float m = gsum[c] * invE;
        float v = gsq[c] * invE - m * m;
        float sc = gamma[c] * rsqrtf(v + 1e-5f);
        scale[c] = sc;
        shift[c] = beta[c] - m * sc;
    }
}

__global__ __launch_bounds__(256) void bn_apply(float* __restrict__ out,
                                                const float* __restrict__ scale,
                                                const float* __restrict__ shift,
                                                long long n4) {
    __shared__ float sc[HDIM], sh[HDIM];
    if (threadIdx.x < HDIM) { sc[threadIdx.x] = scale[threadIdx.x]; sh[threadIdx.x] = shift[threadIdx.x]; }
    __syncthreads();
    long long i = (long long)blockIdx.x * 256 + threadIdx.x;
    long long stride = (long long)gridDim.x * 256;
    for (; i < n4; i += stride) {
        float4* p = (float4*)out + i;
        float4 x = *p;
        int c = (int)((i & 31) * 4);   // (i*4) % 128
        x.x = x.x * sc[c + 0] + sh[c + 0];
        x.y = x.y * sc[c + 1] + sh[c + 1];
        x.z = x.z * sc[c + 2] + sh[c + 2];
        x.w = x.w * sc[c + 3] + sh[c + 3];
        *p = x;
    }
}

extern "C" void kernel_launch(void* const* d_in, const int* in_sizes, int n_in,
                              void* d_out, int out_size, void* d_ws, size_t ws_size,
                              hipStream_t stream) {
    const float* src_na = (const float*)d_in[0];
    const float* dst_na = (const float*)d_in[1];
    const float* ea     = (const float*)d_in[2];
    const int*   eidx   = (const int*)d_in[3];
    const float* W11 = (const float*)d_in[4];
    const float* b11 = (const float*)d_in[5];
    const float* W12 = (const float*)d_in[6];
    const float* b12 = (const float*)d_in[7];
    const float* W13 = (const float*)d_in[8];
    const float* b13 = (const float*)d_in[9];
    const float* gamma = (const float*)d_in[10];
    const float* beta  = (const float*)d_in[11];

    int E = in_sizes[2] / HDIM;

    float*  gsum  = (float*)d_ws;
    float*  gsq   = (float*)((char*)d_ws + 512);
    float*  scale = (float*)((char*)d_ws + 1024);
    float*  shift = (float*)((char*)d_ws + 1536);
    __bf16* wp    = (__bf16*)((char*)d_ws + 2048);

    hipMemsetAsync(d_ws, 0, 1024, stream);
    pack_weights<<<(WTOTAL + 255) / 256, 256, 0, stream>>>(W11, W12, W13, wp);

    int nblk = (E + TM - 1) / TM;
    edge_mlp<<<nblk, 256, 0, stream>>>(src_na, dst_na, ea, eidx, b11, b12, b13, wp,
                                       (float*)d_out, gsum, gsq, E);
    bn_finalize<<<1, 128, 0, stream>>>(gsum, gsq, gamma, beta, scale, shift, E);

    long long n4 = (long long)E * HDIM / 4;
    bn_apply<<<2048, 256, 0, stream>>>((float*)d_out, scale, shift, n4);
}
